// E_GCL_9320079032381
// MI455X (gfx1250) — compile-verified, matrix operand order checked by reading, not run on hardware
//
#include <hip/hip_runtime.h>
#include <hip/hip_bf16.h>
#include <stdint.h>

#define NN 20000
#define EE 640000
#define DD 128
#define HH 128

typedef __bf16 bf16;
typedef bf16 bf16x16 __attribute__((ext_vector_type(16)));
typedef bf16 bf16x8  __attribute__((ext_vector_type(8)));
typedef float f32x8  __attribute__((ext_vector_type(8)));

// Native conversion: lowers to hardware packed cvt on gfx1250.
__device__ __forceinline__ bf16 f2bf(float f) { return (bf16)f; }

__device__ __forceinline__ float silu_f(float x) {
    return x / (1.0f + __expf(-x));
}

__device__ __forceinline__ bf16x16 combine8(bf16x8 lo, bf16x8 hi) {
    bf16x16 r;
#pragma unroll
    for (int i = 0; i < 8; ++i) { r[i] = lo[i]; r[i + 8] = hi[i]; }
    return r;
}

// A-fragment (16x32 bf16): per lane two contiguous 8-elem (16B) runs at
// p and p+16, where p = row + kt*32 + 8*half (CDNA5 16-bit A layout).
// Works for both global bf16 rows and LDS bf16 rows.
__device__ __forceinline__ bf16x16 afrag_bf(const bf16* p) {
    bf16x8 lo = *(const bf16x8*)p;
    bf16x8 hi = *(const bf16x8*)(p + 16);
    return combine8(lo, hi);
}

// A-fragment built from an f32 row (agg path in node kernel only).
__device__ __forceinline__ bf16x16 afrag_f32(const float* __restrict__ p) {
    bf16x16 r;
#pragma unroll
    for (int i = 0; i < 8; ++i) { r[i] = f2bf(p[i]); r[i + 8] = f2bf(p[16 + i]); }
    return r;
}

// B-fragment (32x16 bf16, K x N) from bf16 row-major weight W[N][ldk].
__device__ __forceinline__ bf16x16 bfrag(const bf16* __restrict__ W, int ldk,
                                         int nt, int kt, int lane) {
    const bf16* p = W + (size_t)(nt * 16 + (lane & 15)) * ldk + kt * 32 + 8 * (lane >> 4);
    bf16x8 lo = *(const bf16x8*)p;
    bf16x8 hi = *(const bf16x8*)(p + 16);
    return combine8(lo, hi);
}

__device__ __forceinline__ f32x8 wmma_bf16(bf16x16 a, bf16x16 b, f32x8 c) {
    return __builtin_amdgcn_wmma_f32_16x16x32_bf16(false, a, false, b,
                                                   (short)0, c, false, false);
}

__device__ __forceinline__ f32x8 zero8() {
    f32x8 z = {0.f, 0.f, 0.f, 0.f, 0.f, 0.f, 0.f, 0.f};
    return z;
}

// ---------------- weight conversion to bf16 ----------------
__global__ __launch_bounds__(256) void egcl_prep(
    const float* __restrict__ We1, const float* __restrict__ We2,
    const float* __restrict__ Wn1, const float* __restrict__ Wn2,
    const float* __restrict__ Wc1,
    bf16* __restrict__ We1b, bf16* __restrict__ We2b,
    bf16* __restrict__ Wn1b, bf16* __restrict__ Wn2b,
    bf16* __restrict__ Wc1b)
{
    int t = blockIdx.x * 256 + threadIdx.x;     // 0 .. 128*256-1
    int n = t >> 8, k = t & 255;
    We1b[n * 256 + k] = f2bf(We1[n * 257 + k]); // drop radial col (handled f32)
    Wn1b[n * 256 + k] = f2bf(Wn1[n * 256 + k]);
    if (k < 128) {
        We2b[n * 128 + k] = f2bf(We2[n * 128 + k]);
        Wn2b[n * 128 + k] = f2bf(Wn2[n * 128 + k]);
        Wc1b[n * 128 + k] = f2bf(Wc1[n * 128 + k]);
    }
}

// -------- init: zero agg, seed coord_out, convert h -> bf16 --------
// Grid covers exactly N*128 threads.
__global__ __launch_bounds__(256) void egcl_init(
    const float* __restrict__ h, const float* __restrict__ coord,
    float* __restrict__ agg, bf16* __restrict__ hb,
    float* __restrict__ coordOut)
{
    long t = (long)blockIdx.x * 256 + threadIdx.x;
    agg[t] = 0.0f;
    hb[t]  = f2bf(h[t]);
    if (t < (long)NN * 3) coordOut[t] = coord[t];
}

// ---------------- edge kernel: 1 wave = 16 edges ----------------
__global__ __launch_bounds__(256) void egcl_edge(
    const bf16* __restrict__ hb, const float* __restrict__ coord,
    const int* __restrict__ eidx,
    const bf16* __restrict__ We1b, const float* __restrict__ We1f,
    const float* __restrict__ be1,
    const bf16* __restrict__ We2b, const float* __restrict__ be2,
    const bf16* __restrict__ Wc1b, const float* __restrict__ bc1,
    const float* __restrict__ Wc2,
    float* __restrict__ agg, float* __restrict__ coordOut)
{
    __shared__ __align__(16) bf16 sY[8][16 * HH];   // per-wave activation tile
    __shared__ int   sSrc[8][16];
    __shared__ float sDiff[8][16][3];
    __shared__ float sRad[8][16];

    const int wave = threadIdx.x >> 5;
    const int lane = threadIdx.x & 31;
    const int m    = lane & 15;
    const int half = lane >> 4;
    const int e0   = (blockIdx.x * 8 + wave) * 16;

    // coord_diff / radial / src ids for this 16-edge tile (lanes 0..15)
    if (lane < 16) {
        int e = e0 + lane;
        int s = eidx[e];
        int d = eidx[EE + e];
        float dx = coord[s * 3 + 0] - coord[d * 3 + 0];
        float dy = coord[s * 3 + 1] - coord[d * 3 + 1];
        float dz = coord[s * 3 + 2] - coord[d * 3 + 2];
        sSrc[wave][lane] = s;
        sDiff[wave][lane][0] = dx;
        sDiff[wave][lane][1] = dy;
        sDiff[wave][lane][2] = dz;
        sRad[wave][lane] = dx * dx + dy * dy + dz * dz;
    }
    // same-wave LDS ops are in-order on CDNA5: no barrier needed.

    long srow = (long)eidx[e0 + m] * DD;
    long drow = (long)eidx[EE + e0 + m] * DD;

    // ---- GEMM1: silu([h_src|h_dst|radial] @ We1^T + be1) ----
    f32x8 acc[8];
#pragma unroll
    for (int nt = 0; nt < 8; ++nt) acc[nt] = zero8();

#pragma unroll
    for (int kt = 0; kt < 8; ++kt) {
        const bf16* p = (kt < 4 ? hb + srow + kt * 32
                                : hb + drow + (kt - 4) * 32) + 8 * half;
        bf16x16 a = afrag_bf(p);          // pure loads, no conversion
#pragma unroll
        for (int nt = 0; nt < 8; ++nt) {
            bf16x16 b = bfrag(We1b, 256, nt, kt, lane);
            acc[nt] = wmma_bf16(a, b, acc[nt]);
        }
    }
#pragma unroll
    for (int nt = 0; nt < 8; ++nt) {
        int n = nt * 16 + m;
        float wlast = We1f[n * 257 + 256];   // radial column, f32 exact
        float bias  = be1[n];
#pragma unroll
        for (int v = 0; v < 8; ++v) {
            int row = v + 8 * half;
            float x = acc[nt][v] + sRad[wave][row] * wlast + bias;
            sY[wave][row * HH + n] = f2bf(silu_f(x));
        }
    }

    // ---- GEMM2: edge_feat = silu(Y1 @ We2^T + be2); scatter agg ----
    f32x8 acc2[8];
#pragma unroll
    for (int nt = 0; nt < 8; ++nt) acc2[nt] = zero8();
    const bf16* yrow = &sY[wave][m * HH];
#pragma unroll
    for (int kt = 0; kt < 4; ++kt) {
        bf16x16 a = afrag_bf(yrow + kt * 32 + 8 * half);
#pragma unroll
        for (int nt = 0; nt < 8; ++nt) {
            bf16x16 b = bfrag(We2b, 128, nt, kt, lane);
            acc2[nt] = wmma_bf16(a, b, acc2[nt]);
        }
    }
#pragma unroll
    for (int nt = 0; nt < 8; ++nt) {
        int n = nt * 16 + m;
        float bias = be2[n];
#pragma unroll
        for (int v = 0; v < 8; ++v) {
            int row = v + 8 * half;
            float ef = silu_f(acc2[nt][v] + bias);
            atomicAdd(&agg[(long)sSrc[wave][row] * HH + n], ef);  // segment_sum
            sY[wave][row * HH + n] = f2bf(ef);                    // reuse tile
        }
    }

    // ---- GEMM3: w = silu(edge_feat @ Wc1^T + bc1) @ Wc2^T ----
    f32x8 acc3[8];
#pragma unroll
    for (int nt = 0; nt < 8; ++nt) acc3[nt] = zero8();
#pragma unroll
    for (int kt = 0; kt < 4; ++kt) {
        bf16x16 a = afrag_bf(yrow + kt * 32 + 8 * half);
#pragma unroll
        for (int nt = 0; nt < 8; ++nt) {
            bf16x16 b = bfrag(Wc1b, 128, nt, kt, lane);
            acc3[nt] = wmma_bf16(a, b, acc3[nt]);
        }
    }
    float part[8];
#pragma unroll
    for (int v = 0; v < 8; ++v) part[v] = 0.0f;
#pragma unroll
    for (int nt = 0; nt < 8; ++nt) {
        int n = nt * 16 + m;
        float bias = bc1[n];
        float wc   = Wc2[n];
#pragma unroll
        for (int v = 0; v < 8; ++v)
            part[v] += silu_f(acc3[nt][v] + bias) * wc;
    }
#pragma unroll
    for (int v = 0; v < 8; ++v) {
#pragma unroll
        for (int mask = 1; mask < 16; mask <<= 1)
            part[v] += __shfl_xor(part[v], mask, 32);   // reduce over 16-lane half
    }
    if (m == 0) {
#pragma unroll
        for (int v = 0; v < 8; ++v) {
            int row = v + 8 * half;
            int s   = sSrc[wave][row];
            float w = part[v];
            atomicAdd(&coordOut[s * 3 + 0], sDiff[wave][row][0] * w);
            atomicAdd(&coordOut[s * 3 + 1], sDiff[wave][row][1] * w);
            atomicAdd(&coordOut[s * 3 + 2], sDiff[wave][row][2] * w);
        }
    }
}

// ---------------- node kernel: 1 wave = 16 nodes ----------------
__global__ __launch_bounds__(256) void egcl_node(
    const float* __restrict__ h, const bf16* __restrict__ hb,
    const float* __restrict__ agg,
    const bf16* __restrict__ Wn1b, const float* __restrict__ bn1,
    const bf16* __restrict__ Wn2b, const float* __restrict__ bn2,
    float* __restrict__ hOut)
{
    __shared__ __align__(16) bf16 sY[8][16 * HH];

    const int wave = threadIdx.x >> 5;
    const int lane = threadIdx.x & 31;
    const int m    = lane & 15;
    const int half = lane >> 4;
    const int tile = blockIdx.x * 8 + wave;
    if (tile * 16 >= NN) return;      // no barriers in this kernel

    long hrow = (long)(tile * 16 + m) * DD;

    f32x8 acc[8];
#pragma unroll
    for (int nt = 0; nt < 8; ++nt) acc[nt] = zero8();
#pragma unroll
    for (int kt = 0; kt < 8; ++kt) {
        bf16x16 a;
        if (kt < 4) a = afrag_bf(hb + hrow + kt * 32 + 8 * half);
        else        a = afrag_f32(agg + hrow + (kt - 4) * 32 + 8 * half);
#pragma unroll
        for (int nt = 0; nt < 8; ++nt) {
            bf16x16 b = bfrag(Wn1b, 256, nt, kt, lane);
            acc[nt] = wmma_bf16(a, b, acc[nt]);
        }
    }
#pragma unroll
    for (int nt = 0; nt < 8; ++nt) {
        int n = nt * 16 + m;
        float bias = bn1[n];
#pragma unroll
        for (int v = 0; v < 8; ++v) {
            int row = v + 8 * half;
            sY[wave][row * HH + n] = f2bf(silu_f(acc[nt][v] + bias));
        }
    }

    f32x8 acc2[8];
#pragma unroll
    for (int nt = 0; nt < 8; ++nt) acc2[nt] = zero8();
    const bf16* yrow = &sY[wave][m * HH];
#pragma unroll
    for (int kt = 0; kt < 4; ++kt) {
        bf16x16 a = afrag_bf(yrow + kt * 32 + 8 * half);
#pragma unroll
        for (int nt = 0; nt < 8; ++nt) {
            bf16x16 b = bfrag(Wn2b, 128, nt, kt, lane);
            acc2[nt] = wmma_bf16(a, b, acc2[nt]);
        }
    }
#pragma unroll
    for (int nt = 0; nt < 8; ++nt) {
        int n = nt * 16 + m;
        float bias = bn2[n];
#pragma unroll
        for (int v = 0; v < 8; ++v) {
            int row = v + 8 * half;
            long nd = (long)(tile * 16 + row) * DD;
            hOut[nd + n] = h[nd + n] + acc2[nt][v] + bias;   // residual
        }
    }
}

// ---------------- launch ----------------
extern "C" void kernel_launch(void* const* d_in, const int* in_sizes, int n_in,
                              void* d_out, int out_size, void* d_ws, size_t ws_size,
                              hipStream_t stream) {
    const float* h     = (const float*)d_in[0];
    const float* coord = (const float*)d_in[1];
    const int*   eidx  = (const int*)d_in[2];
    const float* We1   = (const float*)d_in[3];
    const float* be1   = (const float*)d_in[4];
    const float* We2   = (const float*)d_in[5];
    const float* be2   = (const float*)d_in[6];
    const float* Wn1   = (const float*)d_in[7];
    const float* bn1   = (const float*)d_in[8];
    const float* Wn2   = (const float*)d_in[9];
    const float* bn2   = (const float*)d_in[10];
    const float* Wc1   = (const float*)d_in[11];
    const float* bc1   = (const float*)d_in[12];
    const float* Wc2   = (const float*)d_in[13];

    float* hOut     = (float*)d_out;                 // [N, D]
    float* coordOut = (float*)d_out + (long)NN * DD; // [N, 3]

    char* ws = (char*)d_ws;
    float* agg = (float*)ws;                              // N*128 f32   (10,240,000 B)
    bf16*  hb  = (bf16*)(ws + 10240000);                  // N*128 bf16  ( 5,120,000 B)
    bf16* We1b = (bf16*)(ws + 15360000);                  // 128*256 bf16 (65,536 B)
    bf16* We2b = (bf16*)(ws + 15360000 + 65536);          // 128*128
    bf16* Wn1b = (bf16*)(ws + 15360000 + 65536 + 32768);  // 128*256
    bf16* Wn2b = (bf16*)(ws + 15360000 + 131072 + 32768); // 128*128
    bf16* Wc1b = (bf16*)(ws + 15360000 + 131072 + 65536); // 128*128

    egcl_prep<<<128, 256, 0, stream>>>(We1, We2, Wn1, Wn2, Wc1,
                                       We1b, We2b, Wn1b, Wn2b, Wc1b);
    egcl_init<<<(NN * HH) / 256, 256, 0, stream>>>(h, coord, agg, hb, coordOut);
    egcl_edge<<<EE / (8 * 16), 256, 0, stream>>>(hb, coord, eidx,
                                                 We1b, We1, be1, We2b, be2,
                                                 Wc1b, bc1, Wc2, agg, coordOut);
    egcl_node<<<(NN / 16 + 7) / 8, 256, 0, stream>>>(h, hb, agg, Wn1b, bn1,
                                                     Wn2b, bn2, hOut);
}
